// CFCML_20083267076887
// MI455X (gfx1250) — matrix operand order repaired and checked by reading, not compiled, not run on hardware
//
#include <hip/hip_runtime.h>

// ---------------------------------------------------------------------------
// MI455X / gfx1250: 4-directional Mamba 3D block.
// All GEMMs: v_wmma_f32_16x16x32_bf16, one wave per 16x16 tile (wave32).
// Operands pre-converted to bf16 and stored K-contiguous so tile loads are
// 8x dword per lane per operand (contiguous k = 16*half + s per lane), no per-tile cvt.
// Selective scan: 3-pass chunked scan (CHUNK=128, 96 chunks).
// ---------------------------------------------------------------------------

#define DEV __device__ __forceinline__

typedef __attribute__((ext_vector_type(16))) __bf16 v16bf;
typedef __attribute__((ext_vector_type(8)))  float  v8f;

union BFPack { v16bf v; unsigned short u[16]; unsigned w[8]; };

DEV unsigned short f2bf(float f) {
  union { float f; unsigned u; } c; c.f = f;
  unsigned r = c.u + 0x7FFFu + ((c.u >> 16) & 1u);   // round-to-nearest-even
  return (unsigned short)(r >> 16);
}

static constexpr int BB    = 2;
static constexpr int CC    = 64;
static constexpr int LSEQ  = 12288;
static constexpr int DIN   = 128;
static constexpr int NST   = 16;
static constexpr int DTR   = 4;
static constexpr int NPROJ = DTR + 2 * NST;  // 36
static constexpr int NPAD  = 48;             // w_xproj cols padded for 3 tiles
static constexpr int CHUNK = 128;
static constexpr int NCH   = LSEQ / CHUNK;   // 96
static constexpr float EPS = 1e-5f;

// A tile (16x32 bf16, M=r both halves): dword j holds K-pair at
// k0 = 16*half + 2j.  Row memory must be K-contiguous.
DEV void load_a(BFPack& A, const unsigned short* __restrict__ rowp, int half) {
#pragma unroll
  for (int j = 0; j < 8; ++j) {
    int k0 = 16 * half + 2 * j;
    A.w[j] = *(const unsigned*)(rowp + k0);
  }
}
// B tile (32x16 bf16, N=r): dword j holds K-pair at k0 = 16*half + 2j.
// Column memory must be K-contiguous (transposed storage).
DEV void load_b(BFPack& Bt, const unsigned short* __restrict__ colp, int half) {
#pragma unroll
  for (int j = 0; j < 8; ++j) {
    int k0 = 16 * half + 2 * j;
    Bt.w[j] = *(const unsigned*)(colp + k0);
  }
}

// -------------------- prep: convert/transposed-pack weights -----------------
__global__ void k_prep(const float* __restrict__ w_in,  const float* __restrict__ wx,
                       const float* __restrict__ w_out, const float* __restrict__ nin1,
                       const float* __restrict__ nin2,
                       unsigned short* __restrict__ winT0, unsigned short* __restrict__ winT1,
                       unsigned short* __restrict__ wxpT,
                       unsigned short* __restrict__ woutT0, unsigned short* __restrict__ woutT1,
                       unsigned short* __restrict__ nin1b, unsigned short* __restrict__ nin2b) {
  int i = blockIdx.x * blockDim.x + threadIdx.x;
  if (i < 256 * 64) {                       // winT[n][c] = w_in[c,n]
    int n = i / 64, c = i % 64;
    winT0[i] = f2bf(w_in[c * 256 + n]);
    winT1[i] = f2bf(w_in[(63 - c) * 256 + n]);
  }
  if (i < NPAD * DIN) {                     // wxpT[n][k], zero-padded cols
    int n = i / DIN, k = i % DIN;
    wxpT[i] = (n < NPROJ) ? f2bf(wx[k * NPROJ + n]) : (unsigned short)0;
  }
  if (i < CC * DIN) {                       // woutT[n][k] = w_out[k,n]
    int n = i / DIN, k = i % DIN;
    woutT0[i] = f2bf(w_out[k * CC + n]);
    woutT1[i] = f2bf(w_out[k * CC + (63 - n)]);
  }
  if (i < CC * CC) {                        // nin row-major, K already contiguous
    nin1b[i] = f2bf(nin1[i]);
    nin2b[i] = f2bf(nin2[i]);
  }
}

// ----------- prep: x (B,C,L) f32 -> xbf (B,L,C) bf16 (K-contiguous B op) ----
__global__ void k_cvt_x(const float* __restrict__ x, unsigned short* __restrict__ xbf) {
  size_t i = (size_t)blockIdx.x * blockDim.x + threadIdx.x;
  if (i >= (size_t)BB * LSEQ * CC) return;
  int c = (int)(i % CC);
  int l = (int)((i / CC) % LSEQ);
  int b = (int)(i / ((size_t)CC * LSEQ));
  xbf[i] = f2bf(x[((size_t)b * CC + c) * LSEQ + l]);
}

// ------------------------- GEMM 1: nin1 + BN + ReLU -------------------------
__global__ void k_nin1(const unsigned short* __restrict__ xbf,
                       const unsigned short* __restrict__ nwb,
                       const float* __restrict__ g1, const float* __restrict__ b1,
                       const float* __restrict__ m1, const float* __restrict__ v1,
                       float* __restrict__ act, unsigned short* __restrict__ xf) {
  const int wave = (blockIdx.x * blockDim.x + threadIdx.x) >> 5;
  const int lane = threadIdx.x & 31;
  const int tilesL = LSEQ / 16, tilesO = CC / 16;
  const int nt = wave % tilesL;
  const int ot = (wave / tilesL) % tilesO;
  const int b  = wave / (tilesL * tilesO);
  if (b >= BB) return;
  const int half = lane >> 4, r = lane & 15;
  const unsigned short* arow = nwb + (ot * 16 + r) * CC;
  const unsigned short* bcol = xbf + ((size_t)b * LSEQ + nt * 16 + r) * CC;
  v8f acc = {};
#pragma unroll
  for (int kb = 0; kb < 2; ++kb) {
    BFPack A, Bt;
    load_a(A, arow + kb * 32, half);
    load_b(Bt, bcol + kb * 32, half);
    acc = __builtin_amdgcn_wmma_f32_16x16x32_bf16(false, A.v, false, Bt.v,
                                                  (short)0, acc, false, false);
  }
#pragma unroll
  for (int j = 0; j < 8; ++j) {
    int o = ot * 16 + j + 8 * half;
    int l = nt * 16 + r;
    float s   = g1[o] * rsqrtf(v1[o] + EPS);
    float val = fmaxf((acc[j] - m1[o]) * s + b1[o], 0.f);
    act[((size_t)b * CC + o) * LSEQ + l] = val;
    xf[((size_t)b * LSEQ + l) * CC + o]  = f2bf(val);
  }
}

// ------------------------- GEMM 2: xz = xf @ w_in' --------------------------
__global__ void k_xz(const unsigned short* __restrict__ xf,
                     const unsigned short* __restrict__ winT,
                     float* __restrict__ xz) {
  const int wave = (blockIdx.x * blockDim.x + threadIdx.x) >> 5;
  const int lane = threadIdx.x & 31;
  const int tilesN = 256 / 16, tilesM = LSEQ / 16;
  const int nt = wave % tilesN;
  const int mt = (wave / tilesN) % tilesM;
  const int b  = wave / (tilesN * tilesM);
  if (b >= BB) return;
  const int half = lane >> 4, r = lane & 15;
  const unsigned short* arow = xf + ((size_t)b * LSEQ + mt * 16 + r) * CC;
  const unsigned short* bcol = winT + (nt * 16 + r) * CC;
  __builtin_prefetch(arow, 0, 0);
  v8f acc = {};
#pragma unroll
  for (int kb = 0; kb < 2; ++kb) {
    BFPack A, Bt;
    load_a(A, arow + kb * 32, half);
    load_b(Bt, bcol + kb * 32, half);
    acc = __builtin_amdgcn_wmma_f32_16x16x32_bf16(false, A.v, false, Bt.v,
                                                  (short)0, acc, false, false);
  }
#pragma unroll
  for (int j = 0; j < 8; ++j) {
    int l = mt * 16 + j + 8 * half;
    int n = nt * 16 + r;
    xz[((size_t)b * LSEQ + l) * 256 + n] = acc[j];
  }
}

// ----------------- causal depthwise conv (time dir rL) + SiLU ---------------
__global__ void k_conv_silu(const float* __restrict__ xz, const float* __restrict__ cw,
                            const float* __restrict__ cb, int rL,
                            float* __restrict__ xm, unsigned short* __restrict__ xmb) {
  size_t i = (size_t)blockIdx.x * blockDim.x + threadIdx.x;
  if (i >= (size_t)BB * LSEQ * DIN) return;
  int d = (int)(i % DIN);
  int l = (int)((i / DIN) % LSEQ);
  int b = (int)(i / ((size_t)DIN * LSEQ));
  float a = cb[d];
#pragma unroll
  for (int k = 0; k < 4; ++k) {
    int src = rL ? (l + 3 - k) : (l - 3 + k);
    float xv = (src >= 0 && src < LSEQ) ? xz[((size_t)b * LSEQ + src) * 256 + d] : 0.f;
    a += xv * cw[d * 4 + k];
  }
  float s = a / (1.f + __expf(-a));
  xm[i]  = s;
  xmb[i] = f2bf(s);
}

// -------------------- GEMM 3: dbl = xm @ w_xproj (48 pad) -------------------
__global__ void k_xproj(const unsigned short* __restrict__ xmb,
                        const unsigned short* __restrict__ wxpT,
                        float* __restrict__ dbl) {
  const int wave = (blockIdx.x * blockDim.x + threadIdx.x) >> 5;
  const int lane = threadIdx.x & 31;
  const int tilesN = NPAD / 16, tilesM = LSEQ / 16;
  const int nt = wave % tilesN;
  const int mt = (wave / tilesN) % tilesM;
  const int b  = wave / (tilesN * tilesM);
  if (b >= BB) return;
  const int half = lane >> 4, r = lane & 15;
  const unsigned short* arow = xmb + ((size_t)b * LSEQ + mt * 16 + r) * DIN;
  const unsigned short* bcol = wxpT + (nt * 16 + r) * DIN;
  v8f acc = {};
#pragma unroll
  for (int kb = 0; kb < 4; ++kb) {
    BFPack A, Bt;
    load_a(A, arow + kb * 32, half);
    load_b(Bt, bcol + kb * 32, half);
    acc = __builtin_amdgcn_wmma_f32_16x16x32_bf16(false, A.v, false, Bt.v,
                                                  (short)0, acc, false, false);
  }
  const int n = nt * 16 + r;
  if (n < NPROJ)
#pragma unroll
    for (int j = 0; j < 8; ++j) {
      int l = mt * 16 + j + 8 * half;
      dbl[((size_t)b * LSEQ + l) * NPROJ + n] = acc[j];
    }
}

// -------------------- delta = softplus(dt @ w_dt + b_dt) --------------------
__global__ void k_delta(const float* __restrict__ dbl, const float* __restrict__ wdt,
                        const float* __restrict__ bdt, float* __restrict__ delta) {
  size_t i = (size_t)blockIdx.x * blockDim.x + threadIdx.x;
  if (i >= (size_t)BB * LSEQ * DIN) return;
  int d = (int)(i % DIN);
  size_t bl = i / DIN;
  const float* dt = dbl + bl * NPROJ;
  float s = bdt[d];
#pragma unroll
  for (int t = 0; t < DTR; ++t) s += dt[t] * wdt[t * DIN + d];
  delta[i] = (s > 20.f) ? s : __logf(1.f + __expf(s));
}

// --------------------------- scan pass 1: per-chunk -------------------------
__global__ void k_scan1(const float* __restrict__ delta, const float* __restrict__ xm,
                        const float* __restrict__ dbl, const float* __restrict__ A_log,
                        int rL, float* __restrict__ prodA, float* __restrict__ hend) {
  int tid = blockIdx.x * blockDim.x + threadIdx.x;
  if (tid >= BB * DIN * NCH) return;
  int ch = tid % NCH;
  int d  = (tid / NCH) % DIN;
  int b  = tid / (NCH * DIN);
  float nA[NST], P[NST], h[NST];
#pragma unroll
  for (int n = 0; n < NST; ++n) { nA[n] = -__expf(A_log[d * NST + n]); P[n] = 1.f; h[n] = 0.f; }
  for (int t = ch * CHUNK; t < (ch + 1) * CHUNK; ++t) {
    int l = rL ? (LSEQ - 1 - t) : t;
    size_t bl = (size_t)b * LSEQ + l;
    float de = delta[bl * DIN + d];
    float dx = de * xm[bl * DIN + d];
    const float* bm = dbl + bl * NPROJ + DTR;
#pragma unroll
    for (int n = 0; n < NST; ++n) {
      float a = __expf(de * nA[n]);
      P[n] *= a;
      h[n] = a * h[n] + dx * bm[n];
    }
  }
  float* pp = prodA + (size_t)tid * NST;
  float* he = hend  + (size_t)tid * NST;
#pragma unroll
  for (int n = 0; n < NST; ++n) { pp[n] = P[n]; he[n] = h[n]; }
}

// ----------------------- scan pass 2: inter-chunk scan ----------------------
__global__ void k_scan2(const float* __restrict__ prodA, const float* __restrict__ hend,
                        float* __restrict__ carry) {
  int tid = blockIdx.x * blockDim.x + threadIdx.x;
  if (tid >= BB * DIN * NST) return;
  int n = tid % NST;
  int d = (tid / NST) % DIN;
  int b = tid / (NST * DIN);
  size_t base = ((size_t)b * DIN + d) * NCH;
  float c = 0.f;
  for (int ch = 0; ch < NCH; ++ch) {
    size_t idx = (base + ch) * NST + n;
    carry[idx] = c;
    c = prodA[idx] * c + hend[idx];
  }
}

// -------- scan pass 3: replay with carry, y = h.Cm + xm*D, * silu(z) --------
__global__ void k_scan3(const float* __restrict__ delta, const float* __restrict__ xm,
                        const float* __restrict__ dbl, const float* __restrict__ xz,
                        const float* __restrict__ A_log, const float* __restrict__ Dp,
                        const float* __restrict__ carry, int rL,
                        unsigned short* __restrict__ yfb) {
  int tid = blockIdx.x * blockDim.x + threadIdx.x;
  if (tid >= BB * DIN * NCH) return;
  int ch = tid % NCH;
  int d  = (tid / NCH) % DIN;
  int b  = tid / (NCH * DIN);
  float nA[NST], h[NST];
  const float* cr = carry + (size_t)tid * NST;
#pragma unroll
  for (int n = 0; n < NST; ++n) { nA[n] = -__expf(A_log[d * NST + n]); h[n] = cr[n]; }
  float Dv = Dp[d];
  for (int t = ch * CHUNK; t < (ch + 1) * CHUNK; ++t) {
    int l = rL ? (LSEQ - 1 - t) : t;
    size_t bl = (size_t)b * LSEQ + l;
    float de = delta[bl * DIN + d];
    float xv = xm[bl * DIN + d];
    float dx = de * xv;
    const float* bm = dbl + bl * NPROJ + DTR;
    const float* cm = bm + NST;
    float y = 0.f;
#pragma unroll
    for (int n = 0; n < NST; ++n) {
      float a = __expf(de * nA[n]);
      h[n] = a * h[n] + dx * bm[n];
      y += h[n] * cm[n];
    }
    y += xv * Dv;
    float zv = xz[bl * 256 + DIN + d];
    yfb[bl * DIN + d] = f2bf(y * (zv / (1.f + __expf(-zv))));
  }
}

// ------------- GEMM 4: ym (+)= 0.25 * yf @ w_out' (pre-reversed) ------------
__global__ void k_outproj(const unsigned short* __restrict__ yfb,
                          const unsigned short* __restrict__ woutT,
                          int first, float* __restrict__ ym) {
  const int wave = (blockIdx.x * blockDim.x + threadIdx.x) >> 5;
  const int lane = threadIdx.x & 31;
  const int tilesN = CC / 16, tilesM = LSEQ / 16;
  const int nt = wave % tilesN;
  const int mt = (wave / tilesN) % tilesM;
  const int b  = wave / (tilesN * tilesM);
  if (b >= BB) return;
  const int half = lane >> 4, r = lane & 15;
  const unsigned short* arow = yfb + ((size_t)b * LSEQ + mt * 16 + r) * DIN;
  const unsigned short* bcol = woutT + (nt * 16 + r) * DIN;
  v8f acc = {};
#pragma unroll
  for (int kb = 0; kb < 4; ++kb) {
    BFPack A, Bt;
    load_a(A, arow + kb * 32, half);
    load_b(Bt, bcol + kb * 32, half);
    acc = __builtin_amdgcn_wmma_f32_16x16x32_bf16(false, A.v, false, Bt.v,
                                                  (short)0, acc, false, false);
  }
#pragma unroll
  for (int j = 0; j < 8; ++j) {
    int l = mt * 16 + j + 8 * half;
    int n = nt * 16 + r;
    size_t idx = ((size_t)b * LSEQ + l) * CC + n;
    float v = 0.25f * acc[j];
    ym[idx] = first ? v : (ym[idx] + v);
  }
}

// --------- fuse: sum_bf[b,l,c] = bf16(ym[b,l,c] + act[b,c,l]) ---------------
__global__ void k_fuse(const float* __restrict__ ym, const float* __restrict__ act,
                       unsigned short* __restrict__ sum_bf) {
  size_t i = (size_t)blockIdx.x * blockDim.x + threadIdx.x;
  if (i >= (size_t)BB * LSEQ * CC) return;
  int c = (int)(i % CC);
  int l = (int)((i / CC) % LSEQ);
  int b = (int)(i / ((size_t)CC * LSEQ));
  sum_bf[i] = f2bf(ym[i] + act[((size_t)b * CC + c) * LSEQ + l]);
}

// -------------- GEMM 5: out = relu(bn2(nin2_w @ (ym^T + act))) --------------
__global__ void k_nin2(const unsigned short* __restrict__ sum_bf,
                       const unsigned short* __restrict__ nwb,
                       const float* __restrict__ g2, const float* __restrict__ b2,
                       const float* __restrict__ m2, const float* __restrict__ v2,
                       float* __restrict__ out) {
  const int wave = (blockIdx.x * blockDim.x + threadIdx.x) >> 5;
  const int lane = threadIdx.x & 31;
  const int tilesL = LSEQ / 16, tilesO = CC / 16;
  const int nt = wave % tilesL;
  const int ot = (wave / tilesL) % tilesO;
  const int b  = wave / (tilesL * tilesO);
  if (b >= BB) return;
  const int half = lane >> 4, r = lane & 15;
  const unsigned short* arow = nwb + (ot * 16 + r) * CC;
  const unsigned short* bcol = sum_bf + ((size_t)b * LSEQ + nt * 16 + r) * CC;
  v8f acc = {};
#pragma unroll
  for (int kb = 0; kb < 2; ++kb) {
    BFPack A, Bt;
    load_a(A, arow + kb * 32, half);
    load_b(Bt, bcol + kb * 32, half);
    acc = __builtin_amdgcn_wmma_f32_16x16x32_bf16(false, A.v, false, Bt.v,
                                                  (short)0, acc, false, false);
  }
#pragma unroll
  for (int j = 0; j < 8; ++j) {
    int o = ot * 16 + j + 8 * half;
    int l = nt * 16 + r;
    float s = g2[o] * rsqrtf(v2[o] + EPS);
    out[((size_t)b * CC + o) * LSEQ + l] = fmaxf((acc[j] - m2[o]) * s + b2[o], 0.f);
  }
}

// ---------------------------------------------------------------------------
extern "C" void kernel_launch(void* const* d_in, const int* in_sizes, int n_in,
                              void* d_out, int out_size, void* d_ws, size_t ws_size,
                              hipStream_t stream) {
  (void)in_sizes; (void)n_in; (void)out_size; (void)ws_size;
  const float* x      = (const float*)d_in[0];
  const float* nin_w  = (const float*)d_in[1];
  const float* g1     = (const float*)d_in[2];
  const float* b1     = (const float*)d_in[3];
  const float* m1     = (const float*)d_in[4];
  const float* v1     = (const float*)d_in[5];
  const float* nin2_w = (const float*)d_in[6];
  const float* g2     = (const float*)d_in[7];
  const float* b2     = (const float*)d_in[8];
  const float* m2     = (const float*)d_in[9];
  const float* v2     = (const float*)d_in[10];
  const float* w_in   = (const float*)d_in[11];
  const float* conv_w = (const float*)d_in[12];
  const float* conv_b = (const float*)d_in[13];
  const float* w_xprj = (const float*)d_in[14];
  const float* w_dt   = (const float*)d_in[15];
  const float* b_dt   = (const float*)d_in[16];
  const float* A_log  = (const float*)d_in[17];
  const float* D_par  = (const float*)d_in[18];
  const float* w_out  = (const float*)d_in[19];
  float* outp = (float*)d_out;

  char* ws = (char*)d_ws;
  size_t off = 0;
  auto af = [&](size_t n) { float* p = (float*)(ws + off); off = (off + n * 4 + 255) & ~(size_t)255; return p; };
  auto ah = [&](size_t n) { unsigned short* p = (unsigned short*)(ws + off); off = (off + n * 2 + 255) & ~(size_t)255; return p; };

  float*          act    = af((size_t)BB * CC * LSEQ);
  unsigned short* xbf    = ah((size_t)BB * LSEQ * CC);
  unsigned short* xf     = ah((size_t)BB * LSEQ * CC);
  unsigned short* winT0  = ah(256 * 64);
  unsigned short* winT1  = ah(256 * 64);
  unsigned short* wxpT   = ah(NPAD * DIN);
  unsigned short* woutT0 = ah(CC * DIN);
  unsigned short* woutT1 = ah(CC * DIN);
  unsigned short* nin1b  = ah(CC * CC);
  unsigned short* nin2b  = ah(CC * CC);
  float*          xz     = af((size_t)BB * LSEQ * 256);
  float*          xm     = af((size_t)BB * LSEQ * DIN);
  unsigned short* xmb    = ah((size_t)BB * LSEQ * DIN);
  float*          dbl    = af((size_t)BB * LSEQ * NPROJ);
  float*          delta  = af((size_t)BB * LSEQ * DIN);
  float*          prodA  = af((size_t)BB * DIN * NCH * NST);
  float*          hend   = af((size_t)BB * DIN * NCH * NST);
  float*          carry  = af((size_t)BB * DIN * NCH * NST);
  unsigned short* yfb    = ah((size_t)BB * LSEQ * DIN);
  float*          ym     = af((size_t)BB * LSEQ * CC);
  unsigned short* sumb   = ah((size_t)BB * LSEQ * CC);

  const int ew  = (BB * LSEQ * DIN + 255) / 256;
  const int ec  = (BB * LSEQ * CC + 255) / 256;
  const int scg = (BB * DIN * NCH + 255) / 256;

  k_prep<<<(256 * 64 + 255) / 256, 256, 0, stream>>>(w_in, w_xprj, w_out, nin_w, nin2_w,
                                                     winT0, winT1, wxpT, woutT0, woutT1,
                                                     nin1b, nin2b);
  k_cvt_x<<<ec, 256, 0, stream>>>(x, xbf);
  k_nin1<<<(BB * (CC/16) * (LSEQ/16)) / 8, 256, 0, stream>>>(xbf, nin1b, g1, b1, m1, v1, act, xf);

  for (int dir = 0; dir < 4; ++dir) {
    int rC = dir & 1;
    int rL = (dir >> 1) & 1;
    k_xz<<<(BB * (LSEQ/16) * 16) / 8, 256, 0, stream>>>(xf, rC ? winT1 : winT0, xz);
    k_conv_silu<<<ew, 256, 0, stream>>>(xz, conv_w, conv_b, rL, xm, xmb);
    k_xproj<<<(BB * (LSEQ/16) * (NPAD/16)) / 8, 256, 0, stream>>>(xmb, wxpT, dbl);
    k_delta<<<ew, 256, 0, stream>>>(dbl, w_dt, b_dt, delta);
    k_scan1<<<scg, 256, 0, stream>>>(delta, xm, dbl, A_log, rL, prodA, hend);
    k_scan2<<<(BB * DIN * NST + 255) / 256, 256, 0, stream>>>(prodA, hend, carry);
    k_scan3<<<scg, 256, 0, stream>>>(delta, xm, dbl, xz, A_log, D_par, carry, rL, yfb);
    k_outproj<<<(BB * (LSEQ/16) * (CC/16)) / 8, 256, 0, stream>>>(yfb, rC ? woutT1 : woutT0,
                                                                  dir == 0, ym);
  }

  k_fuse<<<ec, 256, 0, stream>>>(ym, act, sumb);
  k_nin2<<<(BB * (CC/16) * (LSEQ/16)) / 8, 256, 0, stream>>>(sumb, nin2b, g2, b2, m2, v2, outp);
}
